// ModalLocalMaskedMHCA_21543555957467
// MI455X (gfx1250) — matrix operand order repaired and checked by reading, not compiled, not run on hardware
//
#include <hip/hip_runtime.h>
#include <hip/hip_bf16.h>
#include <math.h>

typedef __bf16 bf16;
typedef __attribute__((ext_vector_type(16))) __bf16 v16bf;
typedef __attribute__((ext_vector_type(8)))  __bf16 v8bf;
typedef __attribute__((ext_vector_type(4)))  __bf16 v4bf;
typedef __attribute__((ext_vector_type(2)))  __bf16 v2bf;
typedef __attribute__((ext_vector_type(8)))  float  v8f;
typedef __attribute__((ext_vector_type(4)))  float  v4f;

#define C_DIM 512
#define H_DIM 16
#define D_DIM 32
#define ATT_SCALE 0.17677669529663687f   /* 1/sqrt(32) */
#define LN_EPS 1e-5f

// Load a v16bf WMMA operand as two contiguous 16-byte LDS chunks
// (lowers to ds_load_b128 x2).
static __device__ __forceinline__ v16bf lds_frag16(const bf16* p0, const bf16* p1)
{
    v8bf lo = *(const v8bf*)p0;
    v8bf hi = *(const v8bf*)p1;
    return __builtin_shufflevector(lo, hi, 0, 1, 2, 3, 4, 5, 6, 7,
                                           8, 9, 10, 11, 12, 13, 14, 15);
}
// A-matrix fragment (16x32 bf16): K pairs at half*8 and 16+half*8 within row.
static __device__ __forceinline__ v16bf fragA(const bf16* row, int half)
{
    return lds_frag16(row + half * 8, row + 16 + half * 8);
}
// B-matrix fragment (32x16 bf16), N-major storage: K pairs at half*8 and 16+half*8 within row.
static __device__ __forceinline__ v16bf fragB(const bf16* row, int half)
{
    return lds_frag16(row + half * 8, row + 16 + half * 8);
}

// ---------------------------------------------------------------------------
// Fused dwconv3 * mask + channel LayerNorm for the 6 qkv streams.
// grid = (T/16, B, 6), block = 256 (16 t-lanes x 16 channel-groups of 32).
// ---------------------------------------------------------------------------
__global__ __launch_bounds__(256) void qkv_norm_kernel(
    const float* __restrict__ x, const float* __restrict__ xa,
    const float* __restrict__ dw_w, const float* __restrict__ ln_g,
    const float* __restrict__ ln_b, const unsigned char* __restrict__ mask,
    float* __restrict__ Sbase, int T, int B)
{
    __shared__ float ys[C_DIM][17];
    __shared__ float psum[16][17];
    __shared__ float psq[16][17];
    __shared__ float smu[16], srs[16];

    const int s  = blockIdx.z;
    const int b  = blockIdx.y;
    const int t0 = blockIdx.x * 16;
    const int ti = threadIdx.x & 15;
    const int cg = threadIdx.x >> 4;
    const int t  = t0 + ti;

    const float* in = (s == 0 || s == 2 || s == 3) ? x : xa;
    const float mval = mask[b * T + t] ? 1.f : 0.f;

    float sum = 0.f, sq = 0.f;
    for (int j = 0; j < 32; ++j) {
        int c = cg * 32 + j;
        const float* row = in + (size_t)(b * C_DIM + c) * T;
        float xm1 = (t > 0)     ? row[t - 1] : 0.f;
        float x0  = row[t];
        float xp1 = (t < T - 1) ? row[t + 1] : 0.f;
        const float* w = dw_w + (size_t)(s * C_DIM + c) * 3;
        float y = (w[0] * xm1 + w[1] * x0 + w[2] * xp1) * mval;
        ys[c][ti] = y;
        sum += y; sq += y * y;
    }
    psum[cg][ti] = sum; psq[cg][ti] = sq;
    __syncthreads();
    if (cg == 0) {
        float S = 0.f, Q = 0.f;
        for (int g = 0; g < 16; ++g) { S += psum[g][ti]; Q += psq[g][ti]; }
        float mu  = S * (1.f / 512.f);
        float var = Q * (1.f / 512.f) - mu * mu;
        smu[ti] = mu;
        srs[ti] = rsqrtf(var + LN_EPS);
    }
    __syncthreads();
    float mu = smu[ti], rs = srs[ti];
    float* out = Sbase + (size_t)s * B * C_DIM * T + (size_t)b * C_DIM * T;
    for (int j = 0; j < 32; ++j) {
        int c = cg * 32 + j;
        float yn = (ys[c][ti] - mu) * rs * ln_g[s * C_DIM + c] + ln_b[s * C_DIM + c];
        out[(size_t)c * T + t] = yn;
    }
}

// ---------------------------------------------------------------------------
// conv1x1 GEMM on WMMA bf16: Out[b,o,t] = sum_k W[o,k]*In[b,k,t] + bias[o]
// Tile BM=64 x BN=128, BK=32. 8 waves = 4(M) x 2(N); each wave: 1 A-frag
// reused over 4 B-frags -> 4 v_wmma per k-step. Tile fill split into a
// batched global-load phase (6 b128 loads pipelined under one wait) and a
// convert/LDS-store phase.
// mode: 0 = bias, 1 = bias+sigmoid, 2 = bias then * mask[b,t]
// ---------------------------------------------------------------------------
__global__ __launch_bounds__(256) void gemm_conv1x1(
    const float* __restrict__ In, const float* __restrict__ Wt,
    const float* __restrict__ bias, float* __restrict__ Out,
    const unsigned char* __restrict__ mask,
    int K, int T, size_t outBS, int cOff, int mode)
{
    __shared__ __align__(16) bf16 As[64][40];     // [o_local][k]
    __shared__ __align__(16) bf16 BsT[128][40];   // [t_local][k]

    const int tid  = threadIdx.x;
    const int b    = blockIdx.z;
    const int bm   = blockIdx.y * 64;
    const int bn   = blockIdx.x * 128;
    const int lane = tid & 31, wid = tid >> 5;
    const int mg = wid >> 1, ng = wid & 1;
    const int mrow = lane & 15, half = lane >> 4;

    // Per-thread tile-fill coordinates (loop-invariant)
    const int ar0 = tid >> 3,          ac0 = (tid & 7) * 4;          // A chunk 0
    const int ar1 = (tid + 256) >> 3,  ac1 = ac0;                    // A chunk 1
    const int bc0 = (tid & 31) * 4,    br0 = (tid >> 5) * 2;         // B chunk 0
    const int bc1 = bc0,               br1 = ((tid + 256) >> 5) * 2; // B chunk 1

    v8f acc[4] = {{}, {}, {}, {}};
    const float* inB = In + (size_t)b * K * T;

    for (int k0 = 0; k0 < K; k0 += 32) {
        // ---- load phase: issue all 6 global b128 loads back-to-back ----
        v4f wa0 = *(const v4f*)&Wt[(size_t)(bm + ar0) * K + k0 + ac0];
        v4f wa1 = *(const v4f*)&Wt[(size_t)(bm + ar1) * K + k0 + ac1];
        const float* bp0 = &inB[(size_t)(k0 + br0) * T + bn + bc0];
        const float* bp1 = &inB[(size_t)(k0 + br1) * T + bn + bc1];
        v4f ba00 = *(const v4f*)bp0;
        v4f ba01 = *(const v4f*)(bp0 + T);
        v4f ba10 = *(const v4f*)bp1;
        v4f ba11 = *(const v4f*)(bp1 + T);

        // ---- store phase: convert + packed LDS stores ----
        {
            v4bf p;
            p[0] = (bf16)wa0[0]; p[1] = (bf16)wa0[1];
            p[2] = (bf16)wa0[2]; p[3] = (bf16)wa0[3];
            *(v4bf*)&As[ar0][ac0] = p;
            p[0] = (bf16)wa1[0]; p[1] = (bf16)wa1[1];
            p[2] = (bf16)wa1[2]; p[3] = (bf16)wa1[3];
            *(v4bf*)&As[ar1][ac1] = p;
        }
        #pragma unroll
        for (int j = 0; j < 4; ++j) {
            v2bf p; p[0] = (bf16)ba00[j]; p[1] = (bf16)ba01[j];
            *(v2bf*)&BsT[bc0 + j][br0] = p;
        }
        #pragma unroll
        for (int j = 0; j < 4; ++j) {
            v2bf p; p[0] = (bf16)ba10[j]; p[1] = (bf16)ba11[j];
            *(v2bf*)&BsT[bc1 + j][br1] = p;
        }
        if (k0 + 32 < K) {
            __builtin_prefetch(&inB[(size_t)(k0 + 32 + (tid >> 7)) * T + bn + (tid & 127)], 0, 1);
            __builtin_prefetch(&Wt[(size_t)(bm + (tid >> 2)) * K + k0 + 32 + (tid & 3) * 8], 0, 1);
        }
        __syncthreads();

        v16bf a = fragA(&As[mg * 16 + mrow][0], half);
        v16bf bfr[4];
        #pragma unroll
        for (int j = 0; j < 4; ++j)
            bfr[j] = fragB(&BsT[ng * 64 + j * 16 + mrow][0], half);
        #pragma unroll
        for (int j = 0; j < 4; ++j)
            acc[j] = __builtin_amdgcn_wmma_f32_16x16x32_bf16(
                false, a, false, bfr[j], (short)0, acc[j], false, false);
        __syncthreads();
    }

    #pragma unroll
    for (int j = 0; j < 4; ++j) {
        #pragma unroll
        for (int r = 0; r < 8; ++r) {
            int m = mg * 16 + half * 8 + r;
            int o = bm + m;
            int tcol = bn + ng * 64 + j * 16 + mrow;
            float v = acc[j][r] + bias[o];
            if (mode == 1) {
                v = 1.f / (1.f + __expf(-v));
            } else if (mode == 2) {
                v *= mask[b * T + tcol] ? 1.f : 0.f;
            }
            Out[(size_t)b * outBS + (size_t)(cOff + o) * T + tcol] = v;
        }
    }
}

// ---------------------------------------------------------------------------
// Flash-style full cross attention per (b, h, 64-query tile).
// grid = (T/64, H, B), block = 256 (8 waves: 4 Mq x 2 N).
// ---------------------------------------------------------------------------
__global__ __launch_bounds__(256) void cross_attn_kernel(
    const float* __restrict__ Qx, const float* __restrict__ Kx,
    const float* __restrict__ Vx, const unsigned char* __restrict__ mask,
    float* __restrict__ Om, int T)
{
    __shared__ __align__(16) bf16 Qs[64][40];    // [q][d]
    __shared__ __align__(16) bf16 Ks[64][40];    // [s][d]
    __shared__ __align__(16) bf16 VsT[32][72];   // [d][s]
    __shared__ __align__(16) float Ss[64][68];   // scores
    __shared__ __align__(16) bf16 Ps[64][72];    // probabilities
    __shared__ float rowM[64], rowL[64], rowScale[64];
    __shared__ float pmax[64][4], psumS[64][4];
    __shared__ unsigned char mvalid[64];

    const int tid = threadIdx.x;
    const int b = blockIdx.z, h = blockIdx.y;
    const int tq0 = blockIdx.x * 64;
    const int lane = tid & 31, wid = tid >> 5;
    const int mg = wid >> 1, ng = wid & 1;
    const int mrow = lane & 15, half = lane >> 4;
    const int srow = tid >> 2, sq = tid & 3;     // softmax: 4 threads per row

    const float* qb = Qx + (size_t)(b * C_DIM + h * D_DIM) * T;
    const float* kb = Kx + (size_t)(b * C_DIM + h * D_DIM) * T;
    const float* vb = Vx + (size_t)(b * C_DIM + h * D_DIM) * T;

    // tile-fill coordinates
    const int r4 = (tid & 15) * 4, d2 = (tid >> 4) * 2;       // Q/K pair-pack
    const int vd0 = tid >> 4,          vr0 = (tid & 15) * 4;  // V chunk 0
    const int vd1 = (tid + 256) >> 4,  vr1 = vr0;             // V chunk 1

    // Q tile: pair-packed transpose (scaled)
    {
        const float* p0 = &qb[(size_t)d2 * T + tq0 + r4];
        v4f a0 = *(const v4f*)p0;
        v4f a1 = *(const v4f*)(p0 + T);
        #pragma unroll
        for (int j = 0; j < 4; ++j) {
            v2bf p;
            p[0] = (bf16)(a0[j] * ATT_SCALE);
            p[1] = (bf16)(a1[j] * ATT_SCALE);
            *(v2bf*)&Qs[r4 + j][d2] = p;
        }
    }
    if (tid < 64) { rowM[tid] = -1e30f; rowL[tid] = 0.f; }
    v8f acc = {};
    __syncthreads();

    for (int st = 0; st < T; st += 64) {
        // ---- load phase: 4 global b128 loads pipelined ----
        const float* kp = &kb[(size_t)d2 * T + st + r4];
        v4f ka0 = *(const v4f*)kp;
        v4f ka1 = *(const v4f*)(kp + T);
        v4f va0 = *(const v4f*)&vb[(size_t)vd0 * T + st + vr0];
        v4f va1 = *(const v4f*)&vb[(size_t)vd1 * T + st + vr1];

        // ---- store phase ----
        #pragma unroll
        for (int j = 0; j < 4; ++j) {
            v2bf p; p[0] = (bf16)ka0[j]; p[1] = (bf16)ka1[j];
            *(v2bf*)&Ks[r4 + j][d2] = p;
        }
        {
            v4bf p;
            p[0] = (bf16)va0[0]; p[1] = (bf16)va0[1];
            p[2] = (bf16)va0[2]; p[3] = (bf16)va0[3];
            *(v4bf*)&VsT[vd0][vr0] = p;
            p[0] = (bf16)va1[0]; p[1] = (bf16)va1[1];
            p[2] = (bf16)va1[2]; p[3] = (bf16)va1[3];
            *(v4bf*)&VsT[vd1][vr1] = p;
        }
        if (tid < 64) mvalid[tid] = mask[b * T + st + tid];
        __syncthreads();

        // ---- S tile = Q K^T (K-dim = D = 32 -> one WMMA step) ----
        v16bf aq = fragA(&Qs[mg * 16 + mrow][0], half);
        v16bf b0 = fragB(&Ks[ng * 32 + mrow][0], half);
        v16bf b1 = fragB(&Ks[ng * 32 + 16 + mrow][0], half);
        v8f s0 = {}, s1 = {};
        s0 = __builtin_amdgcn_wmma_f32_16x16x32_bf16(false, aq, false, b0, (short)0, s0, false, false);
        s1 = __builtin_amdgcn_wmma_f32_16x16x32_bf16(false, aq, false, b1, (short)0, s1, false, false);
        #pragma unroll
        for (int r = 0; r < 8; ++r) {
            int m = mg * 16 + half * 8 + r;
            Ss[m][ng * 32 + mrow]      = s0[r];
            Ss[m][ng * 32 + 16 + mrow] = s1[r];
        }
        __syncthreads();

        // ---- online softmax: 4 threads per row ----
        {
            float pm = -1e30f;
            #pragma unroll
            for (int j = 0; j < 16; ++j) {
                int col = sq * 16 + j;
                if (mvalid[col]) pm = fmaxf(pm, Ss[srow][col]);
            }
            pmax[srow][sq] = pm;
        }
        __syncthreads();
        if (tid < 64) {
            float mOld = rowM[tid];
            float mNew = fmaxf(fmaxf(pmax[tid][0], pmax[tid][1]),
                               fmaxf(pmax[tid][2], pmax[tid][3]));
            mNew = fmaxf(mOld, mNew);
            float sc = __expf(mOld - mNew);
            rowScale[tid] = sc;
            rowL[tid] *= sc;
            rowM[tid] = mNew;
        }
        __syncthreads();
        {
            float mNew = rowM[srow];
            float ps = 0.f;
            #pragma unroll
            for (int j = 0; j < 16; j += 2) {
                int col = sq * 16 + j;
                float p0 = mvalid[col]     ? __expf(Ss[srow][col] - mNew)     : 0.f;
                float p1 = mvalid[col + 1] ? __expf(Ss[srow][col + 1] - mNew) : 0.f;
                v2bf pp; pp[0] = (bf16)p0; pp[1] = (bf16)p1;
                *(v2bf*)&Ps[srow][col] = pp;
                ps += p0 + p1;
            }
            psumS[srow][sq] = ps;
        }
        __syncthreads();
        if (tid < 64) {
            rowL[tid] += psumS[tid][0] + psumS[tid][1] + psumS[tid][2] + psumS[tid][3];
        }

        // ---- rescale running O, accumulate P·V (two WMMA k-steps) ----
        #pragma unroll
        for (int r = 0; r < 8; ++r) {
            int m = mg * 16 + half * 8 + r;
            acc[r] *= rowScale[m];
        }
        #pragma unroll
        for (int ks = 0; ks < 2; ++ks) {
            v16bf ap = fragA(&Ps[mg * 16 + mrow][ks * 32], half);
            v16bf bv = fragB(&VsT[ng * 16 + mrow][ks * 32], half);
            acc = __builtin_amdgcn_wmma_f32_16x16x32_bf16(
                false, ap, false, bv, (short)0, acc, false, false);
        }
        __syncthreads();
    }

    #pragma unroll
    for (int r = 0; r < 8; ++r) {
        int m = mg * 16 + half * 8 + r;
        float l = rowL[m];
        float inv = (l > 0.f) ? 1.f / l : 0.f;
        int dcol = ng * 16 + mrow;
        Om[(size_t)(b * C_DIM + h * D_DIM + dcol) * T + tq0 + m] = acc[r] * inv;
    }
}

// ---------------------------------------------------------------------------
// Residual gating: qn = q + g*qc ; aqn = aq + (1-g)*ac
// ---------------------------------------------------------------------------
__global__ void residual_gate_kernel(
    const float* __restrict__ q, const float* __restrict__ aq,
    const float* __restrict__ gate, const float* __restrict__ GC,
    float* __restrict__ qn, float* __restrict__ aqn, int CT, int total)
{
    int i = blockIdx.x * 256 + threadIdx.x;
    if (i >= total) return;
    int b = i / CT;
    int ct = i - b * CT;
    float g  = gate[i];
    float qc = GC[(size_t)b * 2 * CT + ct];
    float ac = GC[(size_t)b * 2 * CT + CT + ct];
    qn[i]  = q[i]  + g * qc;
    aqn[i] = aq[i] + (1.f - g) * ac;
}

// ---------------------------------------------------------------------------
// Local windowed attention (window 19). One thread per (b, h, t). Tiny FLOPs.
// ---------------------------------------------------------------------------
__global__ __launch_bounds__(256) void local_attn_kernel(
    const float* __restrict__ Qh, const float* __restrict__ Kh,
    const float* __restrict__ Vh, const unsigned char* __restrict__ mask,
    float* __restrict__ Out, int T, size_t outBS, int cOff, int total)
{
    int gid = blockIdx.x * 256 + threadIdx.x;
    if (gid >= total) return;
    int t = gid % T;
    int hb = gid / T;
    int h = hb % H_DIM;
    int b = hb / H_DIM;
    const float* qb = Qh + (size_t)(b * C_DIM + h * D_DIM) * T;
    const float* kb = Kh + (size_t)(b * C_DIM + h * D_DIM) * T;
    const float* vb = Vh + (size_t)(b * C_DIM + h * D_DIM) * T;

    float q[32];
    #pragma unroll
    for (int d = 0; d < 32; ++d) q[d] = qb[(size_t)d * T + t] * ATT_SCALE;

    float p[19];
    float mx = -1e30f;
    #pragma unroll
    for (int w = 0; w < 19; ++w) {
        int sidx = t - 9 + w;
        bool valid = (sidx >= 0) && (sidx < T) && (mask[b * T + sidx] != 0);
        float dot = -1e30f;
        if (valid) {
            dot = 0.f;
            #pragma unroll
            for (int d = 0; d < 32; ++d) dot += q[d] * kb[(size_t)d * T + sidx];
            mx = fmaxf(mx, dot);
        }
        p[w] = dot;
    }
    float l = 0.f;
    #pragma unroll
    for (int w = 0; w < 19; ++w) {
        float e = (p[w] > -1e29f) ? __expf(p[w] - mx) : 0.f;
        p[w] = e; l += e;
    }
    float inv = (l > 0.f) ? 1.f / l : 0.f;
    float o[32];
    #pragma unroll
    for (int d = 0; d < 32; ++d) o[d] = 0.f;
    #pragma unroll
    for (int w = 0; w < 19; ++w) {
        if (p[w] != 0.f) {
            int sidx = t - 9 + w;
            #pragma unroll
            for (int d = 0; d < 32; ++d) o[d] += p[w] * vb[(size_t)d * T + sidx];
        }
    }
    #pragma unroll
    for (int d = 0; d < 32; ++d)
        Out[(size_t)b * outBS + (size_t)(cOff + h * D_DIM + d) * T + t] = o[d] * inv;
}

// ---------------------------------------------------------------------------
extern "C" void kernel_launch(void* const* d_in, const int* in_sizes, int n_in,
                              void* d_out, int out_size, void* d_ws, size_t ws_size,
                              hipStream_t stream)
{
    const float* x        = (const float*)d_in[0];
    const float* xa       = (const float*)d_in[1];
    const float* dw_w     = (const float*)d_in[2];
    const float* ln_g     = (const float*)d_in[3];
    const float* ln_b     = (const float*)d_in[4];
    const float* pw_w     = (const float*)d_in[5];
    const float* pw_b     = (const float*)d_in[6];
    const float* ca_w     = (const float*)d_in[7];
    const float* ca_b     = (const float*)d_in[8];
    const float* gate_w   = (const float*)d_in[9];
    const float* gate_b   = (const float*)d_in[10];
    const float* concat_w = (const float*)d_in[11];
    const float* concat_b = (const float*)d_in[12];
    const float* proj_w   = (const float*)d_in[13];
    const float* proj_b   = (const float*)d_in[14];
    const unsigned char* mask = (const unsigned char*)d_in[15];

    const int B = 2;
    const int T = in_sizes[0] / (B * C_DIM);
    const size_t CT  = (size_t)C_DIM * T;
    const size_t BCT = (size_t)B * CT;
    const size_t CC  = (size_t)C_DIM * C_DIM;

    float* ws = (float*)d_ws;
    auto slot = [&](int i) { return ws + (size_t)i * BCT; };
    float* GC = slot(11);   // double-width slot (B, 2C, T) occupying slots 11,12

    dim3 blk(256);
    dim3 gemm_grid(T / 128, C_DIM / 64, B);
    dim3 attn_grid(T / 64, H_DIM, B);
    int totalBCT = (int)BCT;
    int totalBHT = B * H_DIM * T;

    // Phase 1: 6 normalized streams -> slots 0..5 (q, aq, k, v, ak, av)
    qkv_norm_kernel<<<dim3(T / 16, B, 6), blk, 0, stream>>>(
        x, xa, dw_w, ln_g, ln_b, mask, ws, T, B);

    // Phase 2a: cross_attn(q=s0, k=ak=s4, v=av=s5) with ca_w[0]
    const float* W0 = ca_w;              const float* c0 = ca_b;
    gemm_conv1x1<<<gemm_grid, blk, 0, stream>>>(slot(4), W0 + 0 * CC, c0 + 0 * C_DIM, slot(6), mask, C_DIM, T, CT, 0, 0); // kx
    gemm_conv1x1<<<gemm_grid, blk, 0, stream>>>(slot(0), W0 + 1 * CC, c0 + 1 * C_DIM, slot(7), mask, C_DIM, T, CT, 0, 0); // qx
    gemm_conv1x1<<<gemm_grid, blk, 0, stream>>>(slot(5), W0 + 2 * CC, c0 + 2 * C_DIM, slot(8), mask, C_DIM, T, CT, 0, 0); // vx
    cross_attn_kernel<<<attn_grid, blk, 0, stream>>>(slot(7), slot(6), slot(8), mask, slot(9), T);
    gemm_conv1x1<<<gemm_grid, blk, 0, stream>>>(slot(9), W0 + 3 * CC, c0 + 3 * C_DIM, GC, mask, C_DIM, T, 2 * CT, 0, 2); // q_cross -> GC[:C]

    // Phase 2b: cross_attn(aq=s1, k=s2, v=s3) with ca_w[1]
    const float* W1 = ca_w + 4 * CC;     const float* c1 = ca_b + 4 * C_DIM;
    gemm_conv1x1<<<gemm_grid, blk, 0, stream>>>(slot(2), W1 + 0 * CC, c1 + 0 * C_DIM, slot(6), mask, C_DIM, T, CT, 0, 0); // kx
    gemm_conv1x1<<<gemm_grid, blk, 0, stream>>>(slot(1), W1 + 1 * CC, c1 + 1 * C_DIM, slot(7), mask, C_DIM, T, CT, 0, 0); // qx
    gemm_conv1x1<<<gemm_grid, blk, 0, stream>>>(slot(3), W1 + 2 * CC, c1 + 2 * C_DIM, slot(8), mask, C_DIM, T, CT, 0, 0); // vx
    cross_attn_kernel<<<attn_grid, blk, 0, stream>>>(slot(7), slot(6), slot(8), mask, slot(9), T);
    gemm_conv1x1<<<gemm_grid, blk, 0, stream>>>(slot(9), W1 + 3 * CC, c1 + 3 * C_DIM, GC, mask, C_DIM, T, 2 * CT, C_DIM, 2); // aq_cross -> GC[C:]

    // Phase 3: gate = sigmoid(conv1x1(GC, gate_w)) -> slot 10
    gemm_conv1x1<<<gemm_grid, blk, 0, stream>>>(GC, gate_w, gate_b, slot(10), mask, 2 * C_DIM, T, CT, 0, 1);

    // Phase 4: residual gating -> qn in s6, aqn in s7
    residual_gate_kernel<<<(totalBCT + 255) / 256, blk, 0, stream>>>(
        slot(0), slot(1), slot(10), GC, slot(6), slot(7), (int)CT, totalBCT);

    // Phase 5: pointwise projections (pw_w order: q,k,v,aq,ak,av)
    gemm_conv1x1<<<gemm_grid, blk, 0, stream>>>(slot(6), pw_w + 0 * CC, pw_b + 0 * C_DIM, slot(0), mask, C_DIM, T, CT, 0, 0); // q2
    gemm_conv1x1<<<gemm_grid, blk, 0, stream>>>(slot(7), pw_w + 3 * CC, pw_b + 3 * C_DIM, slot(1), mask, C_DIM, T, CT, 0, 0); // aq2
    gemm_conv1x1<<<gemm_grid, blk, 0, stream>>>(slot(2), pw_w + 1 * CC, pw_b + 1 * C_DIM, slot(6), mask, C_DIM, T, CT, 0, 0); // k2
    gemm_conv1x1<<<gemm_grid, blk, 0, stream>>>(slot(3), pw_w + 2 * CC, pw_b + 2 * C_DIM, slot(7), mask, C_DIM, T, CT, 0, 0); // v2
    gemm_conv1x1<<<gemm_grid, blk, 0, stream>>>(slot(4), pw_w + 4 * CC, pw_b + 4 * C_DIM, slot(8), mask, C_DIM, T, CT, 0, 0); // ak2
    gemm_conv1x1<<<gemm_grid, blk, 0, stream>>>(slot(5), pw_w + 5 * CC, pw_b + 5 * C_DIM, slot(9), mask, C_DIM, T, CT, 0, 0); // av2

    // Phase 6: local attention -> GC halves (reused as the concat buffer)
    local_attn_kernel<<<(totalBHT + 255) / 256, blk, 0, stream>>>(
        slot(0), slot(6), slot(7), mask, GC, T, 2 * CT, 0, totalBHT);
    local_attn_kernel<<<(totalBHT + 255) / 256, blk, 0, stream>>>(
        slot(1), slot(8), slot(9), mask, GC, T, 2 * CT, C_DIM, totalBHT);

    // Phase 7: concat conv (K=1024) -> s2, then proj (*mask) -> d_out
    gemm_conv1x1<<<gemm_grid, blk, 0, stream>>>(GC, concat_w, concat_b, slot(2), mask, 2 * C_DIM, T, CT, 0, 0);
    gemm_conv1x1<<<gemm_grid, blk, 0, stream>>>(slot(2), proj_w, proj_b, (float*)d_out, mask, C_DIM, T, CT, 0, 2);
}
